// SpatialAttentionModule_48438641164586
// MI455X (gfx1250) — matrix operand order repaired and checked by reading, not compiled, not run on hardware
//
#include <hip/hip_runtime.h>

typedef __bf16 bf16;
typedef __attribute__((ext_vector_type(16))) __bf16 v16bf;
typedef __attribute__((ext_vector_type(8)))  float  v8f;

#define CH 192      // channels F
#define NP 64       // spatial positions N = 8*8
#define XS_S 200    // bf16 stride for [64 x 192] row-major tiles (mult of 8)
#define HT_S 72     // bf16 stride for hx^T [192 x 64]
#define AF_S 68     // f32 stride for logits [64 x 64]
#define AB_S 72     // bf16 stride for probs [64 x 64]

union FragU { v16bf v; uint4 u[2]; };

// A fragment: 16x32 bf16, rows m0..m0+15, K = kb..kb+31 from row-major [*, stride]
__device__ __forceinline__ v16bf load_frag_a(const bf16* base, int stride, int m0, int kb, int lane) {
  int m = lane & 15, h = lane >> 4;
  const bf16* p = base + (m0 + m) * stride + kb + h * 8;
  FragU f;
  f.u[0] = *(const uint4*)(p);        // elems 0..7  : K = kb + h*8 + 0..7
  f.u[1] = *(const uint4*)(p + 16);   // elems 8..15 : K = kb + 16 + h*8 + 0..7
  return f.v;
}

// B fragment: 32x16 bf16, columns n0..n0+15, K = kb..kb+31.
// base is "column-major of B": base[col*stride + k] (i.e. B^T row-major).
__device__ __forceinline__ v16bf load_frag_b(const bf16* base, int stride, int n0, int kb, int lane) {
  int n = lane & 15, h = lane >> 4;
  const bf16* p = base + (n0 + n) * stride + kb + h * 8;
  FragU f;
  f.u[0] = *(const uint4*)(p);        // K = kb + h*8 + 0..7
  f.u[1] = *(const uint4*)(p + 16);   // K = kb + 16 + h*8 + 0..7
  return f.v;
}

__device__ __forceinline__ v8f wmma_bf16(v16bf a, v16bf b, v8f c) {
  return __builtin_amdgcn_wmma_f32_16x16x32_bf16(false, a, false, b, (short)0, c, false, false);
}

// Convert + transpose the four 192x192 fp32 weights into bf16 "B^T" layout:
// wt[i][d][c] = W_i[c][d], so column d of the GEMM B-operand is contiguous.
__global__ void prep_weights_kernel(const float* __restrict__ Wf, const float* __restrict__ Wg,
                                    const float* __restrict__ Wh, const float* __restrict__ Wo,
                                    bf16* __restrict__ wt) {
  int i = blockIdx.y;
  const float* W = (i == 0) ? Wf : (i == 1) ? Wg : (i == 2) ? Wh : Wo;
  for (int t = blockIdx.x * blockDim.x + threadIdx.x; t < CH * CH; t += gridDim.x * blockDim.x) {
    int d = t / CH, c = t - d * CH;
    wt[i * CH * CH + t] = (bf16)W[c * CH + d];
  }
}

__global__ void __launch_bounds__(256)
spatial_attention_kernel(const float* __restrict__ x, const bf16* __restrict__ wt,
                         const float* __restrict__ bf_, const float* __restrict__ bg_,
                         const float* __restrict__ bh_, const float* __restrict__ bo_,
                         const float* __restrict__ gptr, float* __restrict__ out) {
  __shared__ __align__(16) bf16  xs[NP * XS_S];      // x (bf16), later reused as `second`
  __shared__ __align__(16) bf16  fxs[NP * XS_S];     // queries  [N,F]
  __shared__ __align__(16) bf16  gxs[NP * XS_S];     // keys     [N,F] (B^T for QK^T)
  __shared__ __align__(16) bf16  hxT[CH * HT_S];     // values^T [F,N] (B^T for attn@hx)
  __shared__ __align__(16) float attnf[NP * AF_S];   // logits f32
  __shared__ __align__(16) bf16  attnb[NP * AB_S];   // probs bf16

  const int b    = blockIdx.x;
  const int tid  = threadIdx.x;
  const int lane = tid & 31;
  const int w    = tid >> 5;
  const float* xb = x + (size_t)b * (NP * CH);

  // ---- Stage 0: load x[b], convert to bf16 in LDS ----
  for (int i = tid; i < NP * CH; i += 256) {
    int r = i / CH, c = i - r * CH;
    xs[r * XS_S + c] = (bf16)xb[i];
  }
  __syncthreads();

  // ---- Stage 1: fx = xWf+bf, gx = xWg+bg, hx = xWh+bh (hx stored transposed) ----
  for (int t = w; t < 144; t += 8) {            // 3 outputs x (4x12) tiles
    int which = t / 48;
    int rr = t - which * 48;
    int m0 = (rr / 12) * 16, n0 = (rr - (rr / 12) * 12) * 16;
    const bf16* wb = wt + which * (CH * CH);
    v8f acc = {};
#pragma unroll
    for (int kk = 0; kk < 6; kk++) {
      v16bf a  = load_frag_a(xs, XS_S, m0, kk * 32, lane);
      v16bf bm = load_frag_b(wb, CH, n0, kk * 32, lane);
      acc = wmma_bf16(a, bm, acc);
    }
    int n = lane & 15, h = lane >> 4;
    const float* bias = (which == 0) ? bf_ : (which == 1) ? bg_ : bh_;
    float bv = bias[n0 + n];
    if (which == 2) {
#pragma unroll
      for (int r = 0; r < 8; r++)
        hxT[(n0 + n) * HT_S + (m0 + r + h * 8)] = (bf16)(acc[r] + bv);
    } else {
      bf16* dst = (which == 0) ? fxs : gxs;
#pragma unroll
      for (int r = 0; r < 8; r++)
        dst[(m0 + r + h * 8) * XS_S + (n0 + n)] = (bf16)(acc[r] + bv);
    }
  }
  __syncthreads();

  // ---- Stage 2: logits = fx @ gx^T  [64x64] ----
  for (int t = w; t < 16; t += 8) {
    int m0 = (t >> 2) * 16, n0 = (t & 3) * 16;
    v8f acc = {};
#pragma unroll
    for (int kk = 0; kk < 6; kk++) {
      v16bf a  = load_frag_a(fxs, XS_S, m0, kk * 32, lane);
      v16bf bm = load_frag_b(gxs, XS_S, n0, kk * 32, lane);
      acc = wmma_bf16(a, bm, acc);
    }
    int n = lane & 15, h = lane >> 4;
#pragma unroll
    for (int r = 0; r < 8; r++)
      attnf[(m0 + r + h * 8) * AF_S + (n0 + n)] = acc[r];
  }
  __syncthreads();

  // ---- Stage 3: row softmax (4 lanes per row, shfl_xor reduce) ----
  {
    int row = tid >> 2, q = tid & 3;
    const float* lp = attnf + row * AF_S + q * 16;
    float vb[16];
    float mx = -3.402823e38f;
#pragma unroll
    for (int j = 0; j < 16; j++) { vb[j] = lp[j]; mx = fmaxf(mx, vb[j]); }
    mx = fmaxf(mx, __shfl_xor(mx, 1));
    mx = fmaxf(mx, __shfl_xor(mx, 2));
    float s = 0.f;
#pragma unroll
    for (int j = 0; j < 16; j++) { vb[j] = __expf(vb[j] - mx); s += vb[j]; }
    s += __shfl_xor(s, 1);
    s += __shfl_xor(s, 2);
    float inv = 1.0f / s;
    bf16* op = attnb + row * AB_S + q * 16;
#pragma unroll
    for (int j = 0; j < 16; j++) op[j] = (bf16)(vb[j] * inv);
  }
  __syncthreads();

  // ---- Stage 4: second = attn @ hx  -> reuse xs as bf16 [64x192] ----
  bf16* secs = xs;   // x no longer needed in LDS (residual re-read from global)
  for (int t = w; t < 48; t += 8) {
    int m0 = (t / 12) * 16, n0 = (t - (t / 12) * 12) * 16;
    v8f acc = {};
#pragma unroll
    for (int kk = 0; kk < 2; kk++) {
      v16bf a  = load_frag_a(attnb, AB_S, m0, kk * 32, lane);
      v16bf bm = load_frag_b(hxT, HT_S, n0, kk * 32, lane);
      acc = wmma_bf16(a, bm, acc);
    }
    int n = lane & 15, h = lane >> 4;
#pragma unroll
    for (int r = 0; r < 8; r++)
      secs[(m0 + r + h * 8) * XS_S + (n0 + n)] = (bf16)acc[r];
  }
  __syncthreads();

  // ---- Stage 5: out = x + gamma * (second @ Wo + bo) ----
  float gamma = gptr[0];
  const bf16* wo = wt + 3 * (CH * CH);
  for (int t = w; t < 48; t += 8) {
    int m0 = (t / 12) * 16, n0 = (t - (t / 12) * 12) * 16;
    v8f acc = {};
#pragma unroll
    for (int kk = 0; kk < 6; kk++) {
      v16bf a  = load_frag_a(secs, XS_S, m0, kk * 32, lane);
      v16bf bm = load_frag_b(wo, CH, n0, kk * 32, lane);
      acc = wmma_bf16(a, bm, acc);
    }
    int n = lane & 15, h = lane >> 4;
    float bv = bo_[n0 + n];
#pragma unroll
    for (int r = 0; r < 8; r++) {
      int li = (m0 + r + h * 8) * CH + (n0 + n);
      size_t gi = (size_t)b * (NP * CH) + li;
      out[gi] = xb[li] + gamma * (acc[r] + bv);
    }
  }
}

extern "C" void kernel_launch(void* const* d_in, const int* in_sizes, int n_in,
                              void* d_out, int out_size, void* d_ws, size_t ws_size,
                              hipStream_t stream) {
  const float* x    = (const float*)d_in[0];
  const float* Wf   = (const float*)d_in[1];
  const float* bf_  = (const float*)d_in[2];
  const float* Wg   = (const float*)d_in[3];
  const float* bg_  = (const float*)d_in[4];
  const float* Wh   = (const float*)d_in[5];
  const float* bh_  = (const float*)d_in[6];
  const float* Wo   = (const float*)d_in[7];
  const float* bo_  = (const float*)d_in[8];
  const float* gma  = (const float*)d_in[9];
  float* out = (float*)d_out;
  bf16* wt = (bf16*)d_ws;   // 4 * 192*192 bf16 = 294,912 bytes

  int B = in_sizes[0] / (NP * CH);

  prep_weights_kernel<<<dim3(144, 4, 1), 256, 0, stream>>>(Wf, Wg, Wh, Wo, wt);
  spatial_attention_kernel<<<B, 256, 0, stream>>>(x, wt, bf_, bg_, bh_, bo_, gma, out);
}
